// GeneratorAD_6055903887556
// MI455X (gfx1250) — matrix operand order repaired and checked by reading, not compiled, not run on hardware
//
#include <hip/hip_runtime.h>

typedef __attribute__((ext_vector_type(16))) _Float16 v16h;
typedef __attribute__((ext_vector_type(8)))  float    v8f;
typedef __attribute__((ext_vector_type(4)))  float    v4f;
typedef __attribute__((ext_vector_type(8)))  unsigned v8u;

#define SPLITK 8
#define TILE_J 2048   // 2048*5 floats = 40KB LDS (WGP has 320KB)

// ---------------- Stage 0: pack B fragments of (x @ W1) in WMMA order --------
// xwB[chunk][lane][e] (f16): lane -> column n = lane&15 (features 5..15 = 0),
// half e -> K = chunk*32 + (lane>>4)*16 + e. One thread per (chunk,lane):
// writes one contiguous 32B fragment slice.
__global__ void pack_b(const float* __restrict__ x, const float* __restrict__ W1,
                       v16h* __restrict__ xwB, int N) {
  int t = blockIdx.x * blockDim.x + threadIdx.x;   // t in [0, N)
  if (t >= N) return;
  const int lane = t & 31, chunk = t >> 5;
  const int n = lane & 15, hi = lane >> 4;
  float w0 = 0.f, w1 = 0.f;
  if (n < 5) { w0 = W1[n]; w1 = W1[5 + n]; }       // n>=5 -> zero column pad
  v16h b;
#pragma unroll
  for (int e = 0; e < 16; ++e) {
    const int K = chunk * 32 + hi * 16 + e;
    b[e] = (_Float16)(x[K * 2 + 0] * w0 + x[K * 2 + 1] * w1);
  }
  xwB[t] = b;
}

// ---------------- Stage 1: hacc += adj @ xw  via v_wmma_f32_16x16x32_f16 -----
// One wave owns a 16-row tile; split-K over blockIdx.y; f32 atomic reduction.
// adj in {0,1}: f16(1.0)=0x3C00, so int->f16 pair conversion is one integer mad.
__global__ void gcn_wmma(const int* __restrict__ adj, const v16h* __restrict__ xwB,
                         float* __restrict__ hacc, int N) {
  const int lane = threadIdx.x & 31;
  const int wave = threadIdx.x >> 5;
  const int tileM = (blockIdx.x * 8 + wave) * 16;
  const int m  = lane & 15;   // A: row within tile; B/C: column
  const int hi = lane >> 4;   // lane half selects K/M sub-ranges per ISA layout
  const int row = tileM + m;
  const int kspan = N / SPLITK;
  const int kbeg = blockIdx.y * kspan;
  const int kend = kbeg + kspan;
  const int* arow = adj + (size_t)row * N;

  v8f c = {};
  for (int k0 = kbeg; k0 < kend; k0 += 32) {
    // speculative prefetch ~4 chunks ahead (drops silently if past the end)
    __builtin_prefetch(arow + k0 + 128, 0, 3);

    // A 16x32 f16: lane holds M=lane%16; VGPR p: K = hi*16 + p*2 (+0/1)
    v8u au;
#pragma unroll
    for (int p = 0; p < 8; ++p) {
      const int K = (hi << 4) | (p << 1);
      int2 two = *(const int2*)(arow + k0 + K);
      au[p] = (unsigned)two.x * 0x3C00u + (unsigned)two.y * 0x3C000000u;
    }
    v16h a = __builtin_bit_cast(v16h, au);

    // B 32x16 f16: pre-packed fragment, one 32B contiguous load per lane
    v16h b = xwB[(size_t)(k0 >> 5) * 32 + lane];

    c = __builtin_amdgcn_wmma_f32_16x16x32_f16(false, a, false, b, (short)0, c,
                                               false, false);
  }
  // C layout: VGPR r holds (M = tileM + r + 8*hi, N = lane%16)
  if (m < 5) {
#pragma unroll
    for (int r = 0; r < 8; ++r) {
      const int M = tileM + r + 8 * hi;
      atomicAdd(&hacc[(size_t)M * 5 + m], c[r]);
    }
  }
}

// ---------------- Stage 1b: h = relu(hacc + b1) ------------------------------
__global__ void bias_relu(const float* __restrict__ hacc, const float* __restrict__ b1,
                          float* __restrict__ h, int N) {
  int idx = blockIdx.x * blockDim.x + threadIdx.x;
  if (idx >= N * 5) return;
  float v = hacc[idx] + b1[idx % 5];
  h[idx] = v > 0.f ? v : 0.f;
}

// ---------------- Stage 2: h2 = relu(softmax(h h^T) @ (h@W2) + b2) -----------
// Flash-softmax: one wave per row i, j tiled through LDS, online (m,l,acc).
__global__ void attn_h2(const float* __restrict__ h, const float* __restrict__ W2,
                        const float* __restrict__ b2, float* __restrict__ h2, int N) {
  __shared__ float sh[TILE_J * 5];
  const int lane = threadIdx.x & 31;
  const int wave = threadIdx.x >> 5;
  const int i = blockIdx.x * 8 + wave;

  float q0 = h[i * 5 + 0], q1 = h[i * 5 + 1], q2 = h[i * 5 + 2],
        q3 = h[i * 5 + 3], q4 = h[i * 5 + 4];
  float w0 = W2[0], w1 = W2[1], w2 = W2[2], w3 = W2[3], w4 = W2[4];

  float mx = -1e30f, l = 0.f, acc = 0.f;
  for (int t0 = 0; t0 < N; t0 += TILE_J) {
    __syncthreads();
    for (int idx = threadIdx.x; idx < TILE_J * 5; idx += 256)
      sh[idx] = h[(size_t)t0 * 5 + idx];
    __syncthreads();
    for (int j = lane; j < TILE_J; j += 32) {
      const float* hj = &sh[j * 5];
      float s = q0 * hj[0] + q1 * hj[1] + q2 * hj[2] + q3 * hj[3] + q4 * hj[4];
      float z = w0 * hj[0] + w1 * hj[1] + w2 * hj[2] + w3 * hj[3] + w4 * hj[4];
      float mn = fmaxf(mx, s);
      float sc = __expf(mx - mn);
      float e  = __expf(s - mn);
      l   = l * sc + e;
      acc = acc * sc + e * z;
      mx = mn;
    }
  }
  // combine (m,l,acc) across the wave32
#pragma unroll
  for (int off = 16; off >= 1; off >>= 1) {
    float mo = __shfl_xor(mx, off, 32);
    float lo = __shfl_xor(l, off, 32);
    float ao = __shfl_xor(acc, off, 32);
    float mn = fmaxf(mx, mo);
    float s1 = __expf(mx - mn), s2 = __expf(mo - mn);
    l   = l * s1 + lo * s2;
    acc = acc * s1 + ao * s2;
    mx = mn;
  }
  if (lane == 0) {
    float v = acc / l + b2[0];
    h2[i] = v > 0.f ? v : 0.f;
  }
}

// ---------------- Stage 3a: global top-2 of h2 (h2 >= 0) ---------------------
__global__ void top2_kernel(const float* __restrict__ h2, float* __restrict__ top, int N) {
  __shared__ float sm1[256], sm2[256];
  __shared__ int   si[256];
  float m1 = -1.f, m2 = -1.f; int i1 = -1;
  for (int j = threadIdx.x; j < N; j += 256) {
    float v = h2[j];
    if (v > m1)      { m2 = m1; m1 = v; i1 = j; }
    else if (v > m2) { m2 = v; }
  }
  sm1[threadIdx.x] = m1; sm2[threadIdx.x] = m2; si[threadIdx.x] = i1;
  __syncthreads();
  if (threadIdx.x == 0) {
    for (int t = 1; t < 256; ++t) {
      if (sm1[t] > m1) { m2 = fmaxf(m1, sm2[t]); m1 = sm1[t]; i1 = si[t]; }
      else             { m2 = fmaxf(m2, sm1[t]); }
    }
    top[0] = m1; top[1] = m2; ((int*)top)[2] = i1;
  }
}

// ---------------- Stage 3b: per-row max & 1/denominator ----------------------
// s_ij = h2_i*h2_j, with s_ii forced to 0 (score*(1-eye) BEFORE softmax).
// Row max = max(h2_i * max_{j!=i} h2_j, 0), from global top-2.
__global__ void row_denom(const float* __restrict__ h2, const float* __restrict__ top,
                          float* __restrict__ row_m, float* __restrict__ row_inv, int N) {
  __shared__ float sh[8192];   // 32KB: whole h2 vector
  for (int idx = threadIdx.x; idx < N; idx += 256) sh[idx] = h2[idx];
  __syncthreads();
  const int lane = threadIdx.x & 31;
  const int wave = threadIdx.x >> 5;
  const int i = blockIdx.x * 8 + wave;
  const float v = sh[i];
  const float M1 = top[0], M2 = top[1];
  const int idxmax = ((const int*)top)[2];
  const float mrow = fmaxf(v * ((i == idxmax) ? M2 : M1), 0.0f);
  float sum = 0.f;
  for (int j = lane; j < N; j += 32) {
    float s = (j == i) ? 0.0f : v * sh[j];
    sum += __expf(s - mrow);
  }
#pragma unroll
  for (int off = 16; off >= 1; off >>= 1) sum += __shfl_xor(sum, off, 32);
  if (lane == 0) { row_m[i] = mrow; row_inv[i] = 1.0f / sum; }
}

// ---------------- Stage 3c: stream the 268MB output (NT stores) --------------
__global__ void write_out(const float* __restrict__ h2, const float* __restrict__ row_m,
                          const float* __restrict__ row_inv, float* __restrict__ out, int N) {
  const int i = blockIdx.x;
  const float v    = h2[i];
  const float mrow = row_m[i];
  const float inv  = row_inv[i];
  float* orow = out + (size_t)i * N;
  for (int j = threadIdx.x * 4; j < N; j += 256 * 4) {
    v4f hv = *(const v4f*)(h2 + j);
    v4f r;
    r.x = __expf(((j + 0 == i) ? 0.f : v * hv.x) - mrow) * inv;
    r.y = __expf(((j + 1 == i) ? 0.f : v * hv.y) - mrow) * inv;
    r.z = __expf(((j + 2 == i) ? 0.f : v * hv.z) - mrow) * inv;
    r.w = __expf(((j + 3 == i) ? 0.f : v * hv.w) - mrow) * inv;
    __builtin_nontemporal_store(r, (v4f*)(orow + j));
  }
}

// ---------------- Launch ------------------------------------------------------
extern "C" void kernel_launch(void* const* d_in, const int* in_sizes, int n_in,
                              void* d_out, int out_size, void* d_ws, size_t ws_size,
                              hipStream_t stream) {
  const float* x   = (const float*)d_in[0];
  const int*   adj = (const int*)  d_in[1];
  const float* W1  = (const float*)d_in[2];
  const float* b1  = (const float*)d_in[3];
  const float* W2  = (const float*)d_in[4];
  const float* b2  = (const float*)d_in[5];
  float* out = (float*)d_out;
  const int N = in_sizes[0] / 2;   // x is [N,2] -> N = 8192

  // Workspace layout (aliasing; regions dead before reuse):
  //   A: N*32 bytes  -> xwB (f16 B fragments), later reused as h (N*5 f32)
  //   B: N*5 floats  -> hacc, later reused as row_m (N) + row_inv (N)
  //   C: N + 4 floats-> h2, top2 stats
  v16h*  xwB     = (v16h*)d_ws;                          // N/32 chunks * 32 lanes
  float* h       = (float*)d_ws;                         // alias: xwB dead after gcn_wmma
  float* hacc    = (float*)((char*)d_ws + (size_t)N * 32);
  float* row_m   = hacc;                                 // alias: hacc dead after bias_relu
  float* row_inv = hacc + N;
  float* h2      = hacc + (size_t)N * 5;
  float* top     = h2 + N;

  hipMemsetAsync(hacc, 0, (size_t)N * 5 * sizeof(float), stream);

  pack_b<<<(N + 255) / 256, 256, 0, stream>>>(x, W1, xwB, N);

  dim3 g1(N / 16 / 8, SPLITK);                       // 8 waves/block, split-K
  gcn_wmma<<<g1, 256, 0, stream>>>(adj, xwB, hacc, N);

  bias_relu<<<(N * 5 + 255) / 256, 256, 0, stream>>>(hacc, b1, h, N);

  attn_h2<<<N / 8, 256, 0, stream>>>(h, W2, b2, h2, N);

  top2_kernel<<<1, 256, 0, stream>>>(h2, top, N);

  row_denom<<<N / 8, 256, 0, stream>>>(h2, top, row_m, row_inv, N);

  write_out<<<N, 256, 0, stream>>>(h2, row_m, row_inv, out, N);
}
